// Net_J_66649302499334
// MI455X (gfx1250) — matrix operand order repaired and checked by reading, not compiled, not run on hardware
//
#include <hip/hip_runtime.h>

typedef __attribute__((ext_vector_type(16))) _Float16 v16h;
typedef __attribute__((ext_vector_type(8)))  float    v8f;
typedef __attribute__((ext_vector_type(4)))  unsigned int u32x4;
typedef __attribute__((ext_vector_type(8)))  int          i32x8;
typedef __attribute__((ext_vector_type(4)))  int          i32x4;

constexpr int TILE_B  = 128;   // batch rows per block
constexpr int DIN     = 60;    // input features
constexpr int KPAD    = 64;    // padded K for GEMM1
constexpr int NH      = 256;   // hidden centres
constexpr int DOUT    = 360;   // output features
constexpr int DOUT_P  = 368;   // 23 * 16
constexpr int NTILE_O = 23;

// Dynamic LDS layout (bytes). Phase A (centres/x staging) aliases the low part
// of the W2 region, which is only populated in Phase B after a barrier.
constexpr int OFF_W2   = 0;                  // 368*256*2 = 188416
constexpr int OFF_C    = 0;                  // 256*64*2  =  32768 (phase A alias)
constexpr int OFF_X    = 32768;              // 128*64*2  =  16384 (phase A alias)
constexpr int OFF_PHI  = 188416;             // 8*16*256*2 = 65536
constexpr int OFF_X2   = OFF_PHI + 65536;    // 512
constexpr int OFF_C2   = OFF_X2 + 512;       // 1024
constexpr int OFF_S2   = OFF_C2 + 1024;      // 1024
constexpr int OFF_XF32 = OFF_S2 + 1024;      // 128*60*4 = 30720 (TDM landing zone)
constexpr int SMEM_BYTES = OFF_XF32 + TILE_B * DIN * 4;   // 287232 < 320KB/WGP

#if defined(__has_builtin)
#if __has_builtin(__builtin_amdgcn_tensor_load_to_lds) && __has_builtin(__builtin_amdgcn_s_wait_tensorcnt)
#define USE_TDM 1
#endif
#endif

__global__ __launch_bounds__(256, 1)
void rbf_net_kernel(const float* __restrict__ x,
                    const float* __restrict__ centres,
                    const float* __restrict__ sigmas,
                    const float* __restrict__ W2,
                    float* __restrict__ out)
{
    extern __shared__ unsigned char smem[];
    _Float16* sW2  = (_Float16*)(smem + OFF_W2);
    _Float16* sC   = (_Float16*)(smem + OFF_C);
    _Float16* sX   = (_Float16*)(smem + OFF_X);
    _Float16* sPhi = (_Float16*)(smem + OFF_PHI);
    float*    sX2  = (float*)(smem + OFF_X2);
    float*    sC2  = (float*)(smem + OFF_C2);
    float*    sS2  = (float*)(smem + OFF_S2);

    const int tid  = threadIdx.x;
    const int wave = tid >> 5;     // 8 waves (wave32)
    const int lane = tid & 31;
    const int lrow = lane & 15;    // row (A/D) or column (B/D) index within tile
    const int g    = lane >> 4;    // lane half-group

    // ---------- Phase A0: kick off async x-tile DMA (Tensor Data Mover) ----------
#ifdef USE_TDM
    if (wave == 0) {
        const unsigned lds_off = (unsigned)(uintptr_t)(smem + OFF_XF32);
        const unsigned long long ga =
            (unsigned long long)(const void*)(x + (size_t)blockIdx.x * TILE_B * DIN);
        const unsigned NEL = (unsigned)(TILE_B * DIN);          // 7680 f32 elements
        // D# group0: count=1 | lds_addr | global_addr[56:0] | type=2
        u32x4 g0 = { 1u,
                     lds_off,
                     (unsigned)(ga & 0xffffffffu),
                     (unsigned)((ga >> 32) & 0x01ffffffu) | 0x80000000u };
        // D# group1: wg_mask=0, data_size=4B; 1-D tensor: dim0 = tile_dim0 = NEL
        i32x8 g1 = { (int)(2u << 16),                            // data_size=2 (4B)
                     (int)((NEL & 0xffffu) << 16),               // tensor_dim0[15:0]
                     (int)((NEL >> 16) & 0xffffu) | (1 << 16),   // tensor_dim0[31:16], tensor_dim1=1
                     (int)((NEL & 0xffffu) << 16),               // tile_dim0
                     0,                                          // tile_dim1=0 (1-D), tile_dim2=0
                     (int)NEL,                                   // tensor_dim0_stride[31:0]
                     0, 0 };
        i32x4 gz = { 0, 0, 0, 0 };
#if __clang_major__ >= 23
        i32x8 gz8 = { 0, 0, 0, 0, 0, 0, 0, 0 };
        __builtin_amdgcn_tensor_load_to_lds(g0, g1, gz, gz, gz8, 0);
#else
        __builtin_amdgcn_tensor_load_to_lds(g0, g1, gz, gz, 0);
#endif
        __builtin_amdgcn_s_wait_tensorcnt(0);
    }
#endif

    // ---------------- Phase A1: stage centres as f16 + c2 + sigma^2 ----------------
    {   // one centre row per thread
        const int h = tid;
        const float* crow = centres + h * DIN;
        _Float16* dst = sC + h * KPAD;
        float c2 = 0.f;
        #pragma unroll 4
        for (int k = 0; k < DIN; ++k) { float v = crow[k]; c2 += v * v; dst[k] = (_Float16)v; }
        dst[60] = dst[61] = dst[62] = dst[63] = (_Float16)0.f;
        sC2[h] = c2;
        float s = sigmas[h];
        sS2[h] = s * s;
    }
    // ---------------- Phase A2: x tile -> f16 (from TDM LDS or global) ----------------
#ifdef USE_TDM
    __syncthreads();                       // TDM data visible block-wide
    const float* xb = (const float*)(smem + OFF_XF32);
#else
    const float* xb = x + (size_t)blockIdx.x * TILE_B * DIN;
#endif
    for (int i = tid; i < TILE_B * DIN; i += 256) {
        int r = i / DIN, k = i - r * DIN;
        sX[r * KPAD + k] = (_Float16)xb[i];
    }
    for (int i = tid; i < TILE_B * 4; i += 256)
        sX[(i >> 2) * KPAD + 60 + (i & 3)] = (_Float16)0.f;
    __syncthreads();
    if (tid < TILE_B) {   // ||x||^2 per row (from the f16-rounded values the GEMM sees)
        const _Float16* xr = sX + tid * KPAD;
        float s = 0.f;
        #pragma unroll 4
        for (int k = 0; k < DIN; ++k) { float v = (float)xr[k]; s += v * v; }
        sX2[tid] = s;
    }
    __syncthreads();

    // -------- Stage 1: phi[16 x 256] per wave via WMMA f16 (K=64) --------
    const int m0 = wave * 16;                     // local row base of this wave
    v16h a0, a1;                                  // A: x rows, K=0..31 / 32..63
    {
        const _Float16* xr = sX + (m0 + lrow) * KPAD;
        #pragma unroll
        for (int j = 0; j < 8; ++j) {
            a0[j]     = xr[g * 16 + j];           // K = g*16 + j
            a0[8 + j] = xr[g * 16 + 8 + j];       // K = g*16 + 8 + j
            a1[j]     = xr[32 + g * 16 + j];
            a1[8 + j] = xr[32 + g * 16 + 8 + j];
        }
    }
    float x2r[8];                                 // hoisted: loop-invariant over tiles
    #pragma unroll
    for (int r = 0; r < 8; ++r) x2r[r] = sX2[m0 + g * 8 + r];

    _Float16* phiW = sPhi + wave * (16 * NH);
    for (int t = 0; t < 16; ++t) {
        const int h0 = t * 16;
        const _Float16* cb = sC + (h0 + lrow) * KPAD;   // B column n = lrow
        v16h b0, b1;
        #pragma unroll
        for (int j = 0; j < 16; ++j) {
            b0[j] = cb[g * 16 + j];               // K = g*16 + j   (0..31)
            b1[j] = cb[32 + g * 16 + j];          // K = 32 + g*16 + j
        }
        v8f acc = {};
        acc = __builtin_amdgcn_wmma_f32_16x16x32_f16(false, a0, false, b0, (short)0, acc, false, false);
        acc = __builtin_amdgcn_wmma_f32_16x16x32_f16(false, a1, false, b1, (short)0, acc, false, false);

        // fused RBF epilogue: d2 = x2 + c2 - 2*xc ; phi = exp(-s2*d2)
        const int   h  = h0 + lrow;
        const float c2 = sC2[h];
        const float s2 = sS2[h];
        #pragma unroll
        for (int r = 0; r < 8; ++r) {
            const int m = g * 8 + r;              // D: M = r + 8*g
            float d2 = x2r[r] + c2 - 2.0f * acc[r];
            d2 = d2 < 0.f ? 0.f : d2;
            const float phi = __builtin_amdgcn_exp2f(d2 * s2 * -1.4426950408889634f);
            phiW[m * NH + h] = (_Float16)phi;     // row-major, doubles as D->A transpose
        }
    }

    // ---------------- Phase B: stage W2 as f16 (reuses dead LDS) ----------------
    __syncthreads();
    for (int i = tid; i < DOUT * NH; i += 256) sW2[i] = (_Float16)W2[i];
    for (int i = tid; i < (DOUT_P - DOUT) * NH; i += 256) sW2[DOUT * NH + i] = (_Float16)0.f;
    __syncthreads();

    // -------- Stage 2: out = phi @ W2^T, 23 N-tiles x 8 K-steps of WMMA --------
    // A-matrix (this wave's 16 phi rows, K=256) is invariant across all N-tiles:
    // keep it resident in 64 VGPRs instead of re-reading LDS per WMMA.
    const _Float16* phiR = sPhi + wave * (16 * NH) + lrow * NH;
    v16h aP[8];
    #pragma unroll
    for (int s = 0; s < 8; ++s) {
        const int kb = s * 32;
        #pragma unroll
        for (int j = 0; j < 8; ++j) {
            aP[s][j]     = phiR[kb + g * 16 + j];
            aP[s][8 + j] = phiR[kb + g * 16 + 8 + j];
        }
    }

    const size_t rowbase = ((size_t)blockIdx.x * TILE_B + m0 + g * 8) * (size_t)DOUT;
    for (int t = 0; t < NTILE_O; ++t) {
        const int o0 = t * 16;
        const _Float16* wb = sW2 + (o0 + lrow) * NH;              // B column n = lrow
        v8f acc = {};
        #pragma unroll
        for (int s = 0; s < 8; ++s) {
            const int kb = s * 32;
            v16h b;
            #pragma unroll
            for (int j = 0; j < 16; ++j) b[j] = wb[kb + g * 16 + j];
            acc = __builtin_amdgcn_wmma_f32_16x16x32_f16(false, aP[s], false, b, (short)0, acc, false, false);
        }
        // permuted store: o = fp*36 + j*3 + k  ->  column fp*36 + k*12 + j
        const int o = o0 + lrow;
        if (o < DOUT) {
            const int fp  = o / 36;
            const int rem = o - fp * 36;
            const int j12 = rem / 3;
            const int k3  = rem - j12 * 3;
            const int oc  = fp * 36 + k3 * 12 + j12;
            #pragma unroll
            for (int r = 0; r < 8; ++r)
                out[rowbase + (size_t)r * DOUT + oc] = acc[r];
        }
    }
}

extern "C" void kernel_launch(void* const* d_in, const int* in_sizes, int n_in,
                              void* d_out, int out_size, void* d_ws, size_t ws_size,
                              hipStream_t stream) {
    (void)n_in; (void)out_size; (void)d_ws; (void)ws_size;
    const float* x       = (const float*)d_in[0];
    const float* centres = (const float*)d_in[1];
    const float* sigmas  = (const float*)d_in[2];
    const float* W2      = (const float*)d_in[3];
    float* out = (float*)d_out;

    const int B    = in_sizes[0] / DIN;       // 131072
    const int nblk = B / TILE_B;              // 1024

    hipFuncSetAttribute(reinterpret_cast<const void*>(rbf_net_kernel),
                        hipFuncAttributeMaxDynamicSharedMemorySize, SMEM_BYTES);
    rbf_net_kernel<<<nblk, 256, SMEM_BYTES, stream>>>(x, centres, sigmas, W2, out);
}
